// Attention_73478300500671
// MI455X (gfx1250) — compile-verified, matrix operand order checked by reading, not run on hardware
//
#include <hip/hip_runtime.h>
#include <math.h>

// ---------------------------------------------------------------------------
// Types for CDNA5 WMMA (gfx1250, wave32)
// ---------------------------------------------------------------------------
typedef __bf16 bf16;
typedef __attribute__((ext_vector_type(16))) __bf16 v16bf;
typedef __attribute__((ext_vector_type(8)))  float  v8f;

union FragBF {
    v16bf v;
    uint4 q[2];
    bf16  e[16];
};

// D = A x B + C, bf16 inputs, f32 accumulate (v_wmma_f32_16x16x32_bf16)
static __device__ inline v8f wmma_bf16(v16bf a, v16bf b, v8f c) {
    return __builtin_amdgcn_wmma_f32_16x16x32_bf16(
        /*neg_a=*/false, a, /*neg_b=*/false, b,
        /*c_mod=*/(short)0, c, /*reuse_a=*/false, /*reuse_b=*/false);
}

// A/B fragment (16-bit, 16x32): lane holds 8 K-values at kbase and 8 at
// kbase+16, where kbase = kstep*32 + (lane<16 ? 0 : 8).  Row = lane & 15.
static __device__ inline v16bf load_frag_bf16(const bf16* __restrict__ row, int k0) {
    FragBF f;
    f.q[0] = *(const uint4*)(row + k0);
    f.q[1] = *(const uint4*)(row + k0 + 16);
    return f.v;
}

static __device__ inline v16bf load_frag_f32cvt(const float* __restrict__ row,
                                                int k0, bool valid) {
    FragBF f;
    if (valid) {
#pragma unroll
        for (int i = 0; i < 8; ++i) f.e[i]     = (bf16)row[k0 + i];
#pragma unroll
        for (int i = 0; i < 8; ++i) f.e[8 + i] = (bf16)row[k0 + 16 + i];
    } else {
#pragma unroll
        for (int i = 0; i < 16; ++i) f.e[i] = (bf16)0.0f;
    }
    return f.v;
}

static __device__ inline v16bf frag_from_lds_f32(const float* __restrict__ row, int k0) {
    FragBF f;
#pragma unroll
    for (int i = 0; i < 8; ++i) f.e[i]     = (bf16)row[k0 + i];
#pragma unroll
    for (int i = 0; i < 8; ++i) f.e[8 + i] = (bf16)row[k0 + 16 + i];
    return f.v;
}

// Float atomic-max on LDS via signed-max / unsigned-min trick (init to -inf).
static __device__ inline void atomicMaxF(float* a, float v) {
    if (v >= 0.0f) atomicMax((int*)a, __float_as_int(v));
    else           atomicMin((unsigned int*)a, (unsigned int)__float_as_int(v));
}

// ---------------------------------------------------------------------------
// Problem constants
// ---------------------------------------------------------------------------
#define BB 32
#define LL 196
#define LP 208          // L padded to 13*16
#define DD 512
#define HH 8
#define DH 64
#define MP (BB * LP)    // 6656 padded rows total
#define CTX_N (BB * LL * DD)       // 3,211,264
#define LOGIT_N (BB * BB)          // 1,024
#define PROBS_N (BB * LL * LL)     // 1,229,312
#define VT_MPAD 224

// ---------------------------------------------------------------------------
// Kernel 0: zero the atomically-accumulated probs output region
// ---------------------------------------------------------------------------
__global__ void zero_kernel(float* __restrict__ p, int n) {
    int i = blockIdx.x * blockDim.x + threadIdx.x;
    int stride = gridDim.x * blockDim.x;
    for (; i < n; i += stride) p[i] = 0.0f;
}

// ---------------------------------------------------------------------------
// Kernel 1: P = X @ W^T + bias   (M=6656 padded, N=512, K=512), bf16 out.
// For V also writes vT[b][h][d][m] (m padded row of 224) for the PV gemm.
// Block: 256 threads (8 waves) computing a 64x64 tile.  Each wave owns two
// 16x16 output tiles in the SAME row, so the A fragment is loaded once per
// K-step and reused for two WMMAs.
// ---------------------------------------------------------------------------
__global__ void __launch_bounds__(256) proj_kernel(
    const float* __restrict__ X, const float* __restrict__ W,
    const float* __restrict__ bias, bf16* __restrict__ P,
    bf16* __restrict__ vT) {
    const int lane = threadIdx.x & 31;
    const int wave = threadIdx.x >> 5;
    const int half = lane >> 4;
    const int l16  = lane & 15;
    const int blockN = blockIdx.x * 64;
    const int blockM = blockIdx.y * 64;

    const int tm  = wave >> 1;             // 0..3
    const int tn0 = (wave & 1) * 2;        // 0 or 2

    const int mrow  = blockM + tm * 16 + l16;   // padded global row
    const int batch = mrow / LP;
    const int l     = mrow % LP;
    const bool mvalid = (l < LL);
    const float* xrow = X + ((size_t)batch * LL + l) * DD;

    const int n0 = blockN + tn0 * 16 + l16;
    const int n1 = n0 + 16;
    const float* wrow0 = W + (size_t)n0 * DD;
    const float* wrow1 = W + (size_t)n1 * DD;

    v8f c0 = {}, c1 = {};
    for (int kk = 0; kk < 16; ++kk) {
        const int k0 = kk * 32 + half * 8;
        v16bf a  = load_frag_f32cvt(xrow, k0, mvalid);
        v16bf b0 = load_frag_f32cvt(wrow0, k0, true);
        v16bf b1 = load_frag_f32cvt(wrow1, k0, true);
        c0 = wmma_bf16(a, b0, c0);
        c1 = wmma_bf16(a, b1, c1);
    }
    const float bn0 = bias[n0];
    const float bn1 = bias[n1];
#pragma unroll
    for (int r = 0; r < 8; ++r) {
        const int m  = blockM + tm * 16 + r + half * 8;
        const int bm = m / LP;
        const int lm = m % LP;
        const bool ok = (lm < LL);
        const float v0 = ok ? (c0[r] + bn0) : 0.0f;
        const float v1 = ok ? (c1[r] + bn1) : 0.0f;
        P[(size_t)m * DD + n0] = (bf16)v0;
        P[(size_t)m * DD + n1] = (bf16)v1;
        if (vT) {
            const int h0 = n0 >> 6, d0 = n0 & 63;
            const int h1 = n1 >> 6, d1 = n1 & 63;
            vT[(((size_t)bm * HH + h0) * DH + d0) * VT_MPAD + lm] = (bf16)v0;
            vT[(((size_t)bm * HH + h1) * DH + d1) * VT_MPAD + lm] = (bf16)v1;
        }
    }
}

// ---------------------------------------------------------------------------
// Kernel 2: cross-batch retrieval logits. One block per (a,b) pair.
// Each wave owns a full row of 13 output tiles (16 x 208): the A fragment is
// loaded once per K-step and reused by 13 WMMAs.  Streaming row-max/col-max
// reductions in LDS (float atomic max), then summed.
// ---------------------------------------------------------------------------
__global__ void __launch_bounds__(256) retrieve_kernel(
    const bf16* __restrict__ Pq, const bf16* __restrict__ Pk,
    const float* __restrict__ logit_scale, float* __restrict__ out_logits) {
    __shared__ float rowmax[LP];
    __shared__ float colmax[LP];
    __shared__ float sums[2];

    const int a = blockIdx.y, b = blockIdx.x;
    const int tid = threadIdx.x;
    for (int i = tid; i < LP; i += 256) { rowmax[i] = -INFINITY; colmax[i] = -INFINITY; }
    if (tid < 2) sums[tid] = 0.0f;
    __syncthreads();

    const int lane = tid & 31, wave = tid >> 5;
    const int half = lane >> 4, l16 = lane & 15;

    for (int tm = wave; tm < 13; tm += 8) {
        const bf16* arow = Pq + ((size_t)a * LP + tm * 16 + l16) * DD;
        const bf16* bbase = Pk + ((size_t)b * LP + l16) * DD;

        v8f acc[13];
#pragma unroll
        for (int tn = 0; tn < 13; ++tn) acc[tn] = (v8f){};

        for (int kk = 0; kk < 16; ++kk) {
            const int k0 = kk * 32 + half * 8;
            if (kk + 1 < 16)
                __builtin_prefetch(arow + (kk + 1) * 32 + half * 8, 0, 1);
            v16bf av = load_frag_bf16(arow, k0);
#pragma unroll
            for (int tn = 0; tn < 13; ++tn) {
                v16bf bv = load_frag_bf16(bbase + (size_t)tn * 16 * DD, k0);
                acc[tn] = wmma_bf16(av, bv, acc[tn]);
            }
        }

#pragma unroll
        for (int tn = 0; tn < 13; ++tn) {
            const v8f c = acc[tn];
            const int n = tn * 16 + l16;
            const bool nvalid = (n < LL);

            // column max: reduce over this lane's 8 M-rows
            float cm = -INFINITY;
#pragma unroll
            for (int r = 0; r < 8; ++r) {
                const int m = tm * 16 + r + half * 8;
                if (m < LL) cm = fmaxf(cm, c[r]);
            }
            if (nvalid) atomicMaxF(&colmax[n], cm);

            // row max: reduce over the 16 N-lanes of each half, then LDS atomic
#pragma unroll
            for (int r = 0; r < 8; ++r) {
                float v = nvalid ? c[r] : -INFINITY;
                v = fmaxf(v, __shfl_xor(v, 1));
                v = fmaxf(v, __shfl_xor(v, 2));
                v = fmaxf(v, __shfl_xor(v, 4));
                v = fmaxf(v, __shfl_xor(v, 8));
                if (l16 == 0) {
                    const int m = tm * 16 + r + half * 8;
                    if (m < LL) atomicMaxF(&rowmax[m], v);
                }
            }
        }
    }
    __syncthreads();
    if (tid < LL) {
        atomicAdd(&sums[0], rowmax[tid]);
        atomicAdd(&sums[1], colmax[tid]);
    }
    __syncthreads();
    if (tid == 0) {
        const float scale = __expf(logit_scale[0]);
        out_logits[a * BB + b] = scale * (sums[0] + sums[1]) / (2.0f * (float)LL);
    }
}

// ---------------------------------------------------------------------------
// Kernel 3: per-(b,h) self-attention. 4 waves, each owns a 16x224 f32 LDS
// strip: WMMA scores -> softmax in-strip -> probs mean atomics -> WMMA PV.
// Q fragments (K=64 -> 2 frags) and the 7 LDS-probs fragments are hoisted
// and reused across all N tiles.
// ---------------------------------------------------------------------------
__global__ void __launch_bounds__(128) attn_kernel(
    const bf16* __restrict__ Pq, const bf16* __restrict__ Pk,
    const bf16* __restrict__ vT, float* __restrict__ ctx_ws,
    float* __restrict__ out_probs) {
    __shared__ __align__(16) float strip[4][16][VT_MPAD];

    const int h = blockIdx.x, b = blockIdx.y;
    const int lane = threadIdx.x & 31, wave = threadIdx.x >> 5;
    const int half = lane >> 4, l16 = lane & 15;

    const bf16* qbase = Pq + (size_t)b * LP * DD + h * DH;
    const bf16* kbase = Pk + (size_t)b * LP * DD + h * DH;
    const bf16* vtb   = vT + (((size_t)b * HH + h) * DH) * VT_MPAD;
    float (*S)[VT_MPAD] = strip[wave];

    for (int s = wave; s < 13; s += 4) {
        const int l0 = s * 16;

        // ---- scores: 16 x 208 strip, K = 64 (two WMMA steps) ----
        const bf16* arow = qbase + (size_t)(l0 + l16) * DD;
        const v16bf aq0 = load_frag_bf16(arow, half * 8);
        const v16bf aq1 = load_frag_bf16(arow, 32 + half * 8);
        for (int tn = 0; tn < 13; ++tn) {
            const bf16* brow = kbase + (size_t)(tn * 16 + l16) * DD;
            v8f c = {};
            c = wmma_bf16(aq0, load_frag_bf16(brow, half * 8), c);
            c = wmma_bf16(aq1, load_frag_bf16(brow, 32 + half * 8), c);
#pragma unroll
            for (int r = 0; r < 8; ++r)
                S[r + half * 8][tn * 16 + l16] = c[r] * 0.125f;  // 1/sqrt(64)
        }

        // ---- softmax per row, accumulate probs mean ----
        for (int r = 0; r < 16; ++r) {
            const int l = l0 + r;
            if (l >= LL) {
                for (int ci = lane; ci < VT_MPAD; ci += 32) S[r][ci] = 0.0f;
                continue;
            }
            float lm = -INFINITY;
            for (int ci = lane; ci < LL; ci += 32) lm = fmaxf(lm, S[r][ci]);
            for (int off = 16; off > 0; off >>= 1) lm = fmaxf(lm, __shfl_xor(lm, off));
            float ls = 0.0f;
            for (int ci = lane; ci < LL; ci += 32) {
                const float e = __expf(S[r][ci] - lm);
                S[r][ci] = e;
                ls += e;
            }
            for (int off = 16; off > 0; off >>= 1) ls += __shfl_xor(ls, off);
            const float inv = 1.0f / ls;
            float* oprow = out_probs + ((size_t)b * LL + l) * LL;
            for (int ci = lane; ci < LL; ci += 32) {
                const float p = S[r][ci] * inv;
                S[r][ci] = p;
                atomicAdd(oprow + ci, p * 0.125f);   // mean over H=8
            }
            for (int ci = LL + lane; ci < VT_MPAD; ci += 32) S[r][ci] = 0.0f;
        }

        // ---- ctx strip = probs(16x224) @ v(224x64), B from transposed vT ----
        v16bf ap[7];
#pragma unroll
        for (int kk = 0; kk < 7; ++kk)
            ap[kk] = frag_from_lds_f32(&S[l16][0], kk * 32 + half * 8);

        for (int nt = 0; nt < 4; ++nt) {
            const bf16* brow = vtb + (size_t)(nt * 16 + l16) * VT_MPAD;
            v8f c = {};
#pragma unroll
            for (int kk = 0; kk < 7; ++kk)
                c = wmma_bf16(ap[kk], load_frag_bf16(brow, kk * 32 + half * 8), c);
#pragma unroll
            for (int r = 0; r < 8; ++r) {
                const int l = l0 + r + half * 8;
                if (l < LL)
                    ctx_ws[((size_t)b * LL + l) * DD + h * DH + nt * 16 + l16] = c[r];
            }
        }
    }
}

// ---------------------------------------------------------------------------
// Kernel 4: residual + LayerNorm.  One block (128 threads) per (b,l) row.
// ---------------------------------------------------------------------------
__global__ void __launch_bounds__(128) ln_kernel(
    const float* __restrict__ ctx_ws, const float* __restrict__ qs,
    const float* __restrict__ gamma, const float* __restrict__ beta,
    float* __restrict__ out) {
    const int row = blockIdx.x;
    const float* cr = ctx_ws + (size_t)row * DD;
    const float* qr = qs + (size_t)row * DD;
    float x[4], s = 0.0f, ss = 0.0f;
#pragma unroll
    for (int i = 0; i < 4; ++i) {
        const int c = threadIdx.x + i * 128;
        x[i] = cr[c] + qr[c];
        s += x[i];
        ss += x[i] * x[i];
    }
    for (int off = 16; off > 0; off >>= 1) {
        s  += __shfl_xor(s, off);
        ss += __shfl_xor(ss, off);
    }
    __shared__ float ws_s[4], ws_q[4];
    const int wave = threadIdx.x >> 5, lane = threadIdx.x & 31;
    if (lane == 0) { ws_s[wave] = s; ws_q[wave] = ss; }
    __syncthreads();
    s  = ws_s[0] + ws_s[1] + ws_s[2] + ws_s[3];
    ss = ws_q[0] + ws_q[1] + ws_q[2] + ws_q[3];
    const float mu   = s * (1.0f / (float)DD);
    const float var  = ss * (1.0f / (float)DD) - mu * mu;
    const float rstd = rsqrtf(var + 1e-6f);
    float* orow = out + (size_t)row * DD;
#pragma unroll
    for (int i = 0; i < 4; ++i) {
        const int c = threadIdx.x + i * 128;
        orow[c] = (x[i] - mu) * rstd * gamma[c] + beta[c];
    }
}

// ---------------------------------------------------------------------------
// Host launcher
// ---------------------------------------------------------------------------
extern "C" void kernel_launch(void* const* d_in, const int* in_sizes, int n_in,
                              void* d_out, int out_size, void* d_ws, size_t ws_size,
                              hipStream_t stream) {
    (void)in_sizes; (void)n_in; (void)out_size; (void)ws_size;
    const float* q_in  = (const float*)d_in[0];
    const float* k_in  = (const float*)d_in[1];
    const float* v_in  = (const float*)d_in[2];
    const float* Wq    = (const float*)d_in[3];
    const float* bq    = (const float*)d_in[4];
    const float* Wk    = (const float*)d_in[5];
    const float* bk    = (const float*)d_in[6];
    const float* Wv    = (const float*)d_in[7];
    const float* bv    = (const float*)d_in[8];
    const float* gamma = (const float*)d_in[9];
    const float* beta  = (const float*)d_in[10];
    const float* lsc   = (const float*)d_in[11];
    float* out = (float*)d_out;

    // workspace layout (bf16 projections padded to 208 rows/batch, vT, ctx f32)
    bf16* Pq = (bf16*)d_ws;
    bf16* Pk = Pq + (size_t)MP * DD;
    bf16* Pv = Pk + (size_t)MP * DD;
    bf16* vT = Pv + (size_t)MP * DD;
    float* ctx_ws = (float*)(vT + (size_t)BB * HH * DH * VT_MPAD);

    float* out_ctx    = out;
    float* out_logits = out + CTX_N;
    float* out_probs  = out + CTX_N + LOGIT_N;

    zero_kernel<<<1024, 256, 0, stream>>>(out_probs, PROBS_N);
    proj_kernel<<<dim3(DD / 64, MP / 64), 256, 0, stream>>>(q_in, Wq, bq, Pq, nullptr);
    proj_kernel<<<dim3(DD / 64, MP / 64), 256, 0, stream>>>(k_in, Wk, bk, Pk, nullptr);
    proj_kernel<<<dim3(DD / 64, MP / 64), 256, 0, stream>>>(v_in, Wv, bv, Pv, vT);
    retrieve_kernel<<<dim3(BB, BB), 256, 0, stream>>>(Pq, Pk, lsc, out_logits);
    attn_kernel<<<dim3(HH, BB), 128, 0, stream>>>(Pq, Pk, vT, ctx_ws, out_probs);
    ln_kernel<<<BB * LL, 128, 0, stream>>>(ctx_ws, q_in, gamma, beta, out_ctx);
}
